// Heterogeneous_LightGCN_14740327760424
// MI455X (gfx1250) — matrix operand order repaired and checked by reading, not compiled, not run on hardware
//
#include <hip/hip_runtime.h>

#define NP_  110000
#define NT_  20000
#define NG_  10000
#define E_   500000
#define D_   64

typedef __attribute__((ext_vector_type(16))) _Float16 v16h;
typedef __attribute__((ext_vector_type(8)))  float    v8f;

// ---------------- utility kernels ----------------

__global__ void zero_f32(float* __restrict__ p, int n) {
    int i = blockIdx.x * blockDim.x + threadIdx.x;
    if (i < n) p[i] = 0.0f;
}

__global__ void degree_k(const int* __restrict__ dst, float* __restrict__ deg, int e) {
    int i = blockIdx.x * blockDim.x + threadIdx.x;
    if (i < e) atomicAdd(&deg[dst[i]], 1.0f);
}

__global__ void dinv_k(const float* __restrict__ deg, float* __restrict__ dinv, int n) {
    int i = blockIdx.x * blockDim.x + threadIdx.x;
    if (i < n) {
        float d = deg[i];
        dinv[i] = (d > 0.0f) ? rsqrtf(fmaxf(d, 1.0f)) : 0.0f;
    }
}

// One LGConv layer: y[dst] += x[src] * dinv[src]*dinv[dst]
// Thread per (edge, dim-pair): float2 gather + 2 L2-resident f32 atomics.
__global__ void lgconv_k(const int* __restrict__ src, const int* __restrict__ dst,
                         const float* __restrict__ dinv,
                         const float* __restrict__ x, float* __restrict__ y, int e) {
    int tid = blockIdx.x * blockDim.x + threadIdx.x;
    int ed  = tid >> 5;               // 32 threads per edge
    if (ed >= e) return;
    int d2  = (tid & 31) << 1;        // dim pair 0,2,...,62
    int s = src[ed], t = dst[ed];
    float norm = dinv[s] * dinv[t];
    const float2 v = *(const float2*)(x + (size_t)s * D_ + d2);
    float* yp = y + (size_t)t * D_ + d2;
    atomicAdd(yp,     v.x * norm);
    atomicAdd(yp + 1, v.y * norm);
}

// out = (emb + x1 + x2) * (1/3)   [alpha = 1/(L+1), L=2]
__global__ void combine_k(const float* __restrict__ emb, const float* __restrict__ x1,
                          const float* __restrict__ x2, float* __restrict__ out, int n) {
    int i = blockIdx.x * blockDim.x + threadIdx.x;
    if (i < n) out[i] = (emb[i] + x1[i] + x2[i]) * (1.0f / 3.0f);
}

// ---------------- pre-pack B operands into per-lane WMMA layout ----------------
// B (32x16 f16) for chunk c: lane l holds column N=l%16; elems j<8 =
// W[8*(l/16)+j][c*16 + l%16] (K0..15), elems j>=8 zero (K padded 16->32).
// One v16h (32 B) per (chunk, lane); 4 chunks * 32 lanes * 32 B = 4 KB per matrix.
__global__ void pack_B_k(const float* __restrict__ wsrc, const float* __restrict__ wdst,
                         v16h* __restrict__ bsp, v16h* __restrict__ bdp) {
    int tid  = threadIdx.x;       // single block of 128: (chunk, lane)
    int c    = tid >> 5;
    int lane = tid & 31;
    int lr   = lane & 15;
    int kb   = (lane >> 4) * 8;
    v16h bs, bd;
#pragma unroll
    for (int j = 0; j < 16; ++j) {
        bs[j] = (_Float16)((j < 8) ? wsrc[(kb + j) * D_ + c * 16 + lr] : 0.0f);
        bd[j] = (_Float16)((j < 8) ? wdst[(kb + j) * D_ + c * 16 + lr] : 0.0f);
    }
    bsp[c * 32 + lane] = bs;
    bdp[c * 32 + lane] = bd;
}

// ---------------- fused WMMA feature-linear + gather + dot ----------------
// One wave32 handles 16 edges.
//  A (16x32 f16): 16 edges x 16 features, K padded to 32 with zeros.
//    lanes 0-15 row M=lr hold K0..7 in elems 0..7; lanes 16-31 hold K8..15.
//  B: pre-packed fragments loaded as 2x b128 per chunk per matrix.
//  C/D (16x16 f32): VGPR v, lane l -> row M = v + 8*(l/16), col N = l%16.
__global__ __launch_bounds__(256)
void edge_logit_wmma(const int* __restrict__ up_e, const int* __restrict__ ut_e,
                     const int* __restrict__ ug_e,
                     const float* __restrict__ sf, const float* __restrict__ df,
                     const v16h* __restrict__ bsp, const v16h* __restrict__ bdp,
                     const float* __restrict__ bsrc, const float* __restrict__ bdst,
                     const float* __restrict__ p, const float* __restrict__ t,
                     const float* __restrict__ g,
                     float* __restrict__ out, int ntiles) {
    int wave = threadIdx.x >> 5;
    int tile = blockIdx.x * (blockDim.x >> 5) + wave;
    if (tile >= ntiles) return;            // wave-uniform: EXEC stays all-ones below
    int lane = threadIdx.x & 31;
    int half = lane >> 4;
    int lr   = lane & 15;

    // ---- A tiles: src/dst features for 16 edges, f32 -> f16, vectorized ----
    int erow  = tile * 16 + lr;
    int kbase = half * 8;
    const float4* sfr = (const float4*)(sf + (size_t)erow * 16 + kbase);
    const float4* dfr = (const float4*)(df + (size_t)erow * 16 + kbase);
    float4 s0 = sfr[0], s1 = sfr[1];
    float4 d0 = dfr[0], d1 = dfr[1];

    v16h as, ad;
#pragma unroll
    for (int j = 0; j < 16; ++j) { as[j] = (_Float16)0.0f; ad[j] = (_Float16)0.0f; }
    as[0] = (_Float16)s0.x; as[1] = (_Float16)s0.y; as[2] = (_Float16)s0.z; as[3] = (_Float16)s0.w;
    as[4] = (_Float16)s1.x; as[5] = (_Float16)s1.y; as[6] = (_Float16)s1.z; as[7] = (_Float16)s1.w;
    ad[0] = (_Float16)d0.x; ad[1] = (_Float16)d0.y; ad[2] = (_Float16)d0.z; ad[3] = (_Float16)d0.w;
    ad[4] = (_Float16)d1.x; ad[5] = (_Float16)d1.y; ad[6] = (_Float16)d1.z; ad[7] = (_Float16)d1.w;

    // ---- 8 WMMAs: D = A * W_chunk + bias (B pre-packed, L0/L2-hot) ----
    v8f facc_s[4], facc_d[4];
#pragma unroll
    for (int c = 0; c < 4; ++c) {
        v16h Bs = bsp[c * 32 + lane];
        v16h Bd = bdp[c * 32 + lane];
        float bias_s = bsrc[c * 16 + lr];
        float bias_d = bdst[c * 16 + lr];
        v8f Cs, Cd;
#pragma unroll
        for (int v = 0; v < 8; ++v) { Cs[v] = bias_s; Cd[v] = bias_d; }
        facc_s[c] = __builtin_amdgcn_wmma_f32_16x16x32_f16(
            false, as, false, Bs, (short)0, Cs, false, false);
        facc_d[c] = __builtin_amdgcn_wmma_f32_16x16x32_f16(
            false, ad, false, Bd, (short)0, Cd, false, false);
    }

    // ---- gather propagated embeddings at endpoints, dot, half-wave reduce ----
#pragma unroll
    for (int v = 0; v < 8; ++v) {
        int e  = tile * 16 + v + 8 * half;      // C/D layout row for this VGPR slot
        int us = up_e[e],      ud = up_e[E_ + e];
        int ts = ut_e[e],      td = ut_e[E_ + e];
        int gs = ug_e[e],      gd = ug_e[E_ + e];
        float acc = 0.0f;
#pragma unroll
        for (int c = 0; c < 4; ++c) {
            int dim = c * 16 + lr;
            float sg = p[(size_t)us * D_ + dim] + t[(size_t)ts * D_ + dim]
                     + g[(size_t)gs * D_ + dim];
            float dg = p[(size_t)ud * D_ + dim] + t[(size_t)td * D_ + dim]
                     + g[(size_t)gd * D_ + dim];
            float sv = sg * (1.0f / 3.0f) + facc_s[c][v];
            float dv = dg * (1.0f / 3.0f) + facc_d[c][v];
            acc += sv * dv;
        }
        // butterfly over the 16 lanes holding this edge (stays within half)
#pragma unroll
        for (int off = 8; off; off >>= 1) acc += __shfl_xor(acc, off, 32);
        if (lr == 0) out[e] = acc;
    }
}

// ---------------- host orchestration ----------------

extern "C" void kernel_launch(void* const* d_in, const int* in_sizes, int n_in,
                              void* d_out, int out_size, void* d_ws, size_t ws_size,
                              hipStream_t stream) {
    const int*   up_e   = (const int*)d_in[0];   // [2, E]
    const int*   ut_e   = (const int*)d_in[1];
    const int*   ug_e   = (const int*)d_in[2];
    const float* sf     = (const float*)d_in[3]; // [E,16]
    const float* df     = (const float*)d_in[4];
    const float* up_emb = (const float*)d_in[5]; // [NP,64]
    const float* ut_emb = (const float*)d_in[6];
    const float* ug_emb = (const float*)d_in[7];
    const float* wsrc   = (const float*)d_in[8]; // [16,64]
    const float* bsrc   = (const float*)d_in[9];
    const float* wdst   = (const float*)d_in[10];
    const float* bdst   = (const float*)d_in[11];
    float* out = (float*)d_out;

    // Workspace layout (floats), total ~93 MB — all node arrays fit in 192 MB L2.
    float* ws   = (float*)d_ws;
    float* p    = ws;                              // NP*64
    float* t    = p   + (size_t)NP_ * D_;          // NT*64
    float* g    = t   + (size_t)NT_ * D_;          // NG*64
    float* x1   = g   + (size_t)NG_ * D_;          // NP*64 (max graph)
    float* x2   = x1  + (size_t)NP_ * D_;          // NP*64
    float* deg  = x2  + (size_t)NP_ * D_;          // NP
    float* dinv = deg + NP_;                       // NP
    // packed B fragments (32-byte aligned: all prior regions are 32B multiples)
    v16h* bsp   = (v16h*)(dinv + NP_);             // 4*32 v16h = 4 KB
    v16h* bdp   = bsp + 4 * 32;                    // 4 KB

    auto cdiv = [](long long a, long long b) { return (int)((a + b - 1) / b); };

    auto run_graph = [&](const int* edge, const float* emb, float* outb, int N) {
        const int* src = edge;
        const int* dst = edge + E_;
        zero_f32 <<<cdiv(N, 256), 256, 0, stream>>>(deg, N);
        degree_k <<<cdiv(E_, 256), 256, 0, stream>>>(dst, deg, E_);
        dinv_k   <<<cdiv(N, 256), 256, 0, stream>>>(deg, dinv, N);
        zero_f32 <<<cdiv((long long)N * D_, 256), 256, 0, stream>>>(x1, N * D_);
        lgconv_k <<<cdiv((long long)E_ * 32, 256), 256, 0, stream>>>(src, dst, dinv, emb, x1, E_);
        zero_f32 <<<cdiv((long long)N * D_, 256), 256, 0, stream>>>(x2, N * D_);
        lgconv_k <<<cdiv((long long)E_ * 32, 256), 256, 0, stream>>>(src, dst, dinv, x1, x2, E_);
        combine_k<<<cdiv((long long)N * D_, 256), 256, 0, stream>>>(emb, x1, x2, outb, N * D_);
    };

    pack_B_k<<<1, 128, 0, stream>>>(wsrc, wdst, bsp, bdp);

    run_graph(up_e, up_emb, p, NP_);
    run_graph(ut_e, ut_emb, t, NT_);
    run_graph(ug_e, ug_emb, g, NG_);

    int ntiles = E_ / 16;  // 31250, exact
    edge_logit_wmma<<<cdiv(ntiles, 8), 256, 0, stream>>>(
        up_e, ut_e, ug_e, sf, df, bsp, bdp, bsrc, bdst, p, t, g, out, ntiles);
}
